// Attention_65541200937245
// MI455X (gfx1250) — matrix operand order repaired and checked by reading, not compiled, not run on hardware
//
#include <hip/hip_runtime.h>
#include <stdint.h>
#include <math.h>

typedef __attribute__((ext_vector_type(16))) __bf16 v16bf;
typedef __attribute__((ext_vector_type(8)))  __bf16 v8bf;
typedef __attribute__((ext_vector_type(8)))  float  v8f;
typedef __attribute__((ext_vector_type(4)))  unsigned int v4u;
typedef __attribute__((ext_vector_type(8)))  int v8i;
typedef __attribute__((ext_vector_type(4)))  int v4i;

#define D_MODEL 2048
#define NUM_HEADS 16
#define D_KV 128

// Tensor Data Mover availability / arity selection.
#if defined(__gfx1250__) && __has_builtin(__builtin_amdgcn_tensor_load_to_lds)
#define HAVE_TDM 1
#if __has_include(<hip/amd_detail/amd_gfx1250_TDM.h>)
#define TDM_SIX_ARGS 1
#endif
#else
#define HAVE_TDM 0
#endif

// ---------- helpers ----------------------------------------------------

// 16-element bf16 WMMA fragment: 8 elements at p[0..7], 8 at p[gap..gap+7].
// A-fragments and B-fragments: gap=8 (contiguous, K = 16*half + s).
static __device__ __forceinline__ v16bf load_frag(const __bf16* p, int gap) {
  v8bf lo = *(const v8bf*)p;
  v8bf hi = *(const v8bf*)(p + gap);
  return __builtin_shufflevector(lo, hi, 0, 1, 2, 3, 4, 5, 6, 7,
                                 8, 9, 10, 11, 12, 13, 14, 15);
}

static __device__ __forceinline__ v8f wmma_bf16(v16bf a, v16bf b, v8f c) {
  return __builtin_amdgcn_wmma_f32_16x16x32_bf16(false, a, false, b, (short)0, c,
                                                 false, false);
}

#if HAVE_TDM
// 2-D TDM tile load (global -> LDS), D# per CDNA5 ISA 8.3/8.4:
// group0: [1:0]=count=1, [63:32]=lds_addr, [120:64]=global_addr, [127:126]=type=2
// group1: [17:16]=data_size(1 => 2B), tensor_dim0/1, tile_dim0/1, dim0 stride.
static __device__ __forceinline__ void tdm_load_2d(unsigned lds_off, const void* gptr,
                                                   unsigned tdim0, unsigned tdim1,
                                                   unsigned tile0, unsigned tile1,
                                                   unsigned long long stride0) {
  unsigned long long ga = (unsigned long long)(uintptr_t)gptr;
  v4u g0;
  g0[0] = 1u;                                                   // count=1, user mode
  g0[1] = lds_off;                                              // LDS byte address
  g0[2] = (unsigned)ga;                                         // global_addr[31:0]
  g0[3] = (unsigned)((ga >> 32) & 0x01FFFFFFull) | (2u << 30);  // [56:32] + type=2
  v8i g1;
  g1[0] = (int)(1u << 16);                                      // data_size=2B, mask=0
  g1[1] = (int)((tdim0 & 0xFFFFu) << 16);                       // tensor_dim0 lo
  g1[2] = (int)((tdim0 >> 16) | ((tdim1 & 0xFFFFu) << 16));     // dim0 hi | dim1 lo
  g1[3] = (int)((tdim1 >> 16) | (tile0 << 16));                 // dim1 hi | tile_dim0
  g1[4] = (int)(tile1 & 0xFFFFu);                               // tile_dim1, tile_dim2=0
  g1[5] = (int)(unsigned)stride0;                               // dim0_stride[31:0]
  g1[6] = (int)((stride0 >> 32) & 0xFFFFull);                   // dim0_stride[47:32]
  g1[7] = 0;
  v4i gz = 0;
#if defined(TDM_SIX_ARGS)
  v8i gz8 = 0;
  __builtin_amdgcn_tensor_load_to_lds(g0, g1, gz, gz, gz8, 0);
#else
  __builtin_amdgcn_tensor_load_to_lds(g0, g1, gz, gz, 0);
#endif
}
#endif

// ---------- f32 -> bf16 cast -------------------------------------------

__global__ void cast_bf16_kernel(const float* __restrict__ in, __bf16* __restrict__ out,
                                 long n) {
  long i = (long)blockIdx.x * blockDim.x + threadIdx.x;
  long stride = (long)gridDim.x * blockDim.x;
  for (; i < n; i += stride) out[i] = (__bf16)in[i];
}

// ---------- weight transpose-cast: f32 [K,N] -> bf16 [N,K] -------------

__global__ void wtrans_kernel(const float* __restrict__ W, __bf16* __restrict__ Wt,
                              long total) {
  long idx = (long)blockIdx.x * blockDim.x + threadIdx.x;
  long stride = (long)gridDim.x * blockDim.x;
  for (; idx < total; idx += stride) {
    long k = idx >> 11;             // / D_MODEL
    int n = (int)(idx & 2047);
    Wt[(size_t)n * D_MODEL + k] = (__bf16)W[idx];
  }
}

// ---------- bf16 GEMM (TN): C[M,N] f32 = A[M,K] * Bt[N,K]^T ------------
// grid = (M/128, N/128), block = 256 threads (8 waves).
// Wave w computes a 32x64 sub-tile: rows (w>>1)*32, cols (w&1)*64 -> 8 WMMAs/K-step.
// A/B panels (128x32 bf16) double-buffered in LDS; the Tensor Data Mover prefetches
// panel k+1 while WMMAs consume panel k (s_wait_tensorcnt(1) pipelining).

#define BM 128
#define BN 128
#define BK 32
#define PANEL (BM * BK)
__global__ void __launch_bounds__(256)
gemm_bf16_tn_kernel(const __bf16* __restrict__ A, const __bf16* __restrict__ Bt,
                    float* __restrict__ C, int M, int N, int K) {
  __shared__ __attribute__((aligned(16))) __bf16 Als[2][PANEL];
  __shared__ __attribute__((aligned(16))) __bf16 Bls[2][PANEL];
  int wave_s = __builtin_amdgcn_readfirstlane((int)(threadIdx.x >> 5));
  int lane = threadIdx.x & 31;
  int l16 = lane & 15, hlf = lane >> 4, akb = hlf * 8;
  int wr = wave_s >> 1;             // 0..3 : 32-row strip
  int wc = wave_s & 1;              // 0..1 : 64-col strip
  int m0 = blockIdx.x * BM;
  int n0 = blockIdx.y * BN;
  v8f acc[2][4] = {};

  // Prologue: fill buffer 0.
#if HAVE_TDM
  if (wave_s == 0)
    tdm_load_2d((unsigned)(uintptr_t)&Als[0][0], A + (size_t)m0 * K,
                (unsigned)K, (unsigned)M, BK, BM, (unsigned long long)K);
  else if (wave_s == 1)
    tdm_load_2d((unsigned)(uintptr_t)&Bls[0][0], Bt + (size_t)n0 * K,
                (unsigned)K, (unsigned)N, BK, BN, (unsigned long long)K);
#else
  for (int seg = threadIdx.x; seg < PANEL / 8; seg += 256) {
    int row = seg >> 2, col = (seg & 3) * 8;
    *(v8bf*)&Als[0][row * BK + col] = *(const v8bf*)&A[(size_t)(m0 + row) * K + col];
    *(v8bf*)&Bls[0][row * BK + col] = *(const v8bf*)&Bt[(size_t)(n0 + row) * K + col];
  }
#endif

  int ib = 0;
  for (int k0 = 0; k0 < K; k0 += BK, ib ^= 1) {
    int kn = k0 + BK;
#if HAVE_TDM
    // Prefetch next panels into the other buffer, then wait for the current
    // panel (oldest of <=2 in-flight TDM ops; per-wave TDM ops are in-order).
    if (wave_s == 0) {
      if (kn < K) {
        tdm_load_2d((unsigned)(uintptr_t)&Als[ib ^ 1][0], A + (size_t)m0 * K + kn,
                    (unsigned)K, (unsigned)M, BK, BM, (unsigned long long)K);
        __builtin_amdgcn_s_wait_tensorcnt(1);
      } else {
        __builtin_amdgcn_s_wait_tensorcnt(0);
      }
    } else if (wave_s == 1) {
      if (kn < K) {
        tdm_load_2d((unsigned)(uintptr_t)&Bls[ib ^ 1][0], Bt + (size_t)n0 * K + kn,
                    (unsigned)K, (unsigned)N, BK, BN, (unsigned long long)K);
        __builtin_amdgcn_s_wait_tensorcnt(1);
      } else {
        __builtin_amdgcn_s_wait_tensorcnt(0);
      }
    }
#else
    if (kn < K) {
      for (int seg = threadIdx.x; seg < PANEL / 8; seg += 256) {
        int row = seg >> 2, col = (seg & 3) * 8;
        *(v8bf*)&Als[ib ^ 1][row * BK + col] =
            *(const v8bf*)&A[(size_t)(m0 + row) * K + kn + col];
        *(v8bf*)&Bls[ib ^ 1][row * BK + col] =
            *(const v8bf*)&Bt[(size_t)(n0 + row) * K + kn + col];
      }
    }
#endif
    __syncthreads();
    v16bf af[2], bfr[4];
#pragma unroll
    for (int t = 0; t < 2; ++t)
      af[t] = load_frag(&Als[ib][(wr * 32 + t * 16 + l16) * BK + hlf * 16], 8);
#pragma unroll
    for (int u = 0; u < 4; ++u)
      bfr[u] = load_frag(&Bls[ib][(wc * 64 + u * 16 + l16) * BK + hlf * 16], 8);
#pragma unroll
    for (int t = 0; t < 2; ++t)
#pragma unroll
      for (int u = 0; u < 4; ++u)
        acc[t][u] = wmma_bf16(af[t], bfr[u], acc[t][u]);
    __syncthreads();
  }
#pragma unroll
  for (int t = 0; t < 2; ++t)
#pragma unroll
    for (int u = 0; u < 4; ++u)
#pragma unroll
      for (int r = 0; r < 8; ++r)
        C[(size_t)(m0 + wr * 32 + t * 16 + r + hlf * 8) * N +
          n0 + wc * 64 + u * 16 + l16] = acc[t][u][r];
}

// ---------- RoPE + cast: f32 [B*S, 2048] -> bf16 [B,H,S,128] -----------
// premul folds the attention score scale (1/sqrt(d)) into Q.

__global__ void rope_cast_kernel(const float* __restrict__ In, __bf16* __restrict__ Out,
                                 int S, long total, float premul) {
  long idx = (long)blockIdx.x * blockDim.x + threadIdx.x;
  long stride = (long)gridDim.x * blockDim.x;
  for (; idx < total; idx += stride) {
    long row = idx >> 10;           // / (H*64)
    int rem = (int)(idx & 1023);
    int h = rem >> 6;
    int i = rem & 63;
    int b = (int)(row / S);
    int s = (int)(row % S);
    const float* src = In + row * D_MODEL + h * D_KV + 2 * i;
    float x1 = src[0], x2 = src[1];
    float ang = (float)s * __powf(10000.0f, -(float)i * (1.0f / 64.0f));
    float sn, cs;
    __sincosf(ang, &sn, &cs);
    __bf16* dst = Out + ((size_t)(b * NUM_HEADS + h) * S + s) * D_KV + 2 * i;
    dst[0] = (__bf16)((x1 * cs - x2 * sn) * premul);
    dst[1] = (__bf16)((x1 * sn + x2 * cs) * premul);
  }
}

// ---------- V transpose: f32 [B*S, 2048] -> bf16 [B,H,128,S] -----------

__global__ void vtrans_kernel(const float* __restrict__ V, __bf16* __restrict__ Vt,
                              int S, long total) {
  long idx = (long)blockIdx.x * blockDim.x + threadIdx.x;
  long stride = (long)gridDim.x * blockDim.x;
  for (; idx < total; idx += stride) {
    long row = idx >> 11;           // / D_MODEL
    int col = (int)(idx & 2047);
    int b = (int)(row / S);
    int s = (int)(row % S);
    int h = col >> 7;
    int d = col & 127;
    Vt[((size_t)(b * NUM_HEADS + h) * D_KV + d) * S + s] = (__bf16)V[idx];
  }
}

// ---------- flash attention: one wave per (b, h, 16-query tile) --------
// Qb (pre-scaled)/Kb: bf16 [B,H,S,128]; Vt: bf16 [B,H,128,S]; Ob: bf16 [B*S, 2048].

__global__ void __launch_bounds__(32)
attn_fwd_kernel(const __bf16* __restrict__ Qb, const __bf16* __restrict__ Kb,
                const __bf16* __restrict__ Vt, __bf16* __restrict__ Ob, int S) {
  __shared__ __attribute__((aligned(16))) __bf16 Pld[16 * 32];
  const int qtiles = S / 16;
  int bid = blockIdx.x;
  int qt = bid % qtiles;
  int h  = (bid / qtiles) % NUM_HEADS;
  int b  = bid / (qtiles * NUM_HEADS);
  int q0 = qt * 16;
  int lane = threadIdx.x;
  int l16  = lane & 15;
  int hlf  = lane >> 4;
  int akb  = hlf * 8;
  const __bf16* Qrow = Qb + ((size_t)(b * NUM_HEADS + h) * S) * D_KV;
  const __bf16* Krow = Kb + ((size_t)(b * NUM_HEADS + h) * S) * D_KV;
  const __bf16* Vhd  = Vt + ((size_t)(b * NUM_HEADS + h) * D_KV) * S;

  v16bf aQ[4];
#pragma unroll
  for (int d = 0; d < 4; ++d)
    aQ[d] = load_frag(Qrow + (size_t)(q0 + l16) * D_KV + d * 32 + hlf * 16, 8);

  float rmax[8], rsum[8];
  v8f o[8] = {};
#pragma unroll
  for (int r = 0; r < 8; ++r) { rmax[r] = -1e30f; rsum[r] = 0.0f; }

  int nchunks = q0 / 32 + 1;                 // covers causal k <= q0+15
  for (int c = 0; c < nchunks; ++c) {
    int k0 = c * 32;
    v8f s0 = {}, s1 = {};
    const __bf16* kr0 = Krow + (size_t)(k0 + l16) * D_KV + hlf * 16;
    const __bf16* kr1 = Krow + (size_t)(k0 + 16 + l16) * D_KV + hlf * 16;
#pragma unroll
    for (int d = 0; d < 4; ++d) {
      s0 = wmma_bf16(aQ[d], load_frag(kr0 + d * 32, 8), s0);
      s1 = wmma_bf16(aQ[d], load_frag(kr1 + d * 32, 8), s1);
    }
#pragma unroll
    for (int r = 0; r < 8; ++r) {
      int q = q0 + r + hlf * 8;
      float v0 = s0[r]; if (k0 + l16      > q) v0 = -1e30f;
      float v1 = s1[r]; if (k0 + 16 + l16 > q) v1 = -1e30f;
      float cm = fmaxf(v0, v1);
#pragma unroll
      for (int m = 1; m <= 8; m <<= 1) cm = fmaxf(cm, __shfl_xor(cm, m, 32));
      float nm = fmaxf(rmax[r], cm);
      float corr = __expf(rmax[r] - nm);
      float p0 = __expf(v0 - nm);
      float p1 = __expf(v1 - nm);
      float ps = p0 + p1;
#pragma unroll
      for (int m = 1; m <= 8; m <<= 1) ps += __shfl_xor(ps, m, 32);
      rsum[r] = rsum[r] * corr + ps;
      rmax[r] = nm;
#pragma unroll
      for (int j = 0; j < 8; ++j) o[j][r] *= corr;
      Pld[(r + hlf * 8) * 32 + l16]      = (__bf16)p0;
      Pld[(r + hlf * 8) * 32 + 16 + l16] = (__bf16)p1;
    }
    __syncthreads();
    v16bf aP = load_frag(&Pld[l16 * 32 + hlf * 16], 8);
#pragma unroll
    for (int j = 0; j < 8; ++j) {
      const __bf16* vr = Vhd + (size_t)(j * 16 + l16) * S + k0 + hlf * 16;
      o[j] = wmma_bf16(aP, load_frag(vr, 8), o[j]);
    }
    __syncthreads();
  }
#pragma unroll
  for (int r = 0; r < 8; ++r) {
    float inv = 1.0f / rsum[r];
    size_t row = (size_t)b * S + q0 + r + hlf * 8;
#pragma unroll
    for (int j = 0; j < 8; ++j)
      Ob[row * D_MODEL + h * D_KV + j * 16 + l16] = (__bf16)(o[j][r] * inv);
  }
}

// ---------- launch -----------------------------------------------------

extern "C" void kernel_launch(void* const* d_in, const int* in_sizes, int n_in,
                              void* d_out, int out_size, void* d_ws, size_t ws_size,
                              hipStream_t stream) {
  (void)n_in; (void)out_size; (void)ws_size;
  const float* x  = (const float*)d_in[0];
  const float* Wq = (const float*)d_in[1];
  const float* Wk = (const float*)d_in[2];
  const float* Wv = (const float*)d_in[3];
  const float* Wo = (const float*)d_in[4];
  int  S  = in_sizes[5];
  long BS = (long)in_sizes[0] / D_MODEL;   // B * S
  int  Bb = (int)(BS / S);

  char* ws = (char*)d_ws;
  size_t off = 0;
  auto alloc = [&](size_t bytes) -> void* {
    void* p = ws + off;
    off += (bytes + 255) & ~(size_t)255;
    return p;
  };
  const size_t DD = (size_t)D_MODEL * D_MODEL;
  __bf16* xb  = (__bf16*)alloc((size_t)BS * D_MODEL * 2);
  __bf16* Wqt = (__bf16*)alloc(DD * 2);
  __bf16* Wkt = (__bf16*)alloc(DD * 2);
  __bf16* Wvt = (__bf16*)alloc(DD * 2);
  __bf16* Wot = (__bf16*)alloc(DD * 2);
  float*  Qf  = (float*)alloc((size_t)BS * D_MODEL * 4);
  float*  Kf  = (float*)alloc((size_t)BS * D_MODEL * 4);
  float*  Vf  = (float*)alloc((size_t)BS * D_MODEL * 4);
  __bf16* Qh  = (__bf16*)alloc((size_t)BS * D_MODEL * 2);
  __bf16* Kh  = (__bf16*)alloc((size_t)BS * D_MODEL * 2);
  __bf16* Vt  = (__bf16*)alloc((size_t)BS * D_MODEL * 2);
  __bf16* Ab  = (__bf16*)alloc((size_t)BS * D_MODEL * 2);

  cast_bf16_kernel<<<1024, 256, 0, stream>>>(x, xb, BS * D_MODEL);
  wtrans_kernel<<<512, 256, 0, stream>>>(Wq, Wqt, (long)DD);
  wtrans_kernel<<<512, 256, 0, stream>>>(Wk, Wkt, (long)DD);
  wtrans_kernel<<<512, 256, 0, stream>>>(Wv, Wvt, (long)DD);
  wtrans_kernel<<<512, 256, 0, stream>>>(Wo, Wot, (long)DD);

  dim3 gg((unsigned)(BS / BM), D_MODEL / BN);
  gemm_bf16_tn_kernel<<<gg, 256, 0, stream>>>(xb, Wqt, Qf, (int)BS, D_MODEL, D_MODEL);
  gemm_bf16_tn_kernel<<<gg, 256, 0, stream>>>(xb, Wkt, Kf, (int)BS, D_MODEL, D_MODEL);
  gemm_bf16_tn_kernel<<<gg, 256, 0, stream>>>(xb, Wvt, Vf, (int)BS, D_MODEL, D_MODEL);

  long npair = BS * (D_MODEL / 2);   // BS * H * 64
  const float scale = 0.08838834764831845f;  // 1/sqrt(128)
  rope_cast_kernel<<<4096, 256, 0, stream>>>(Qf, Qh, S, npair, scale);
  rope_cast_kernel<<<4096, 256, 0, stream>>>(Kf, Kh, S, npair, 1.0f);
  vtrans_kernel<<<8192, 256, 0, stream>>>(Vf, Vt, S, BS * D_MODEL);

  attn_fwd_kernel<<<Bb * NUM_HEADS * (S / 16), 32, 0, stream>>>(Qh, Kh, Vt, Ab, S);

  gemm_bf16_tn_kernel<<<gg, 256, 0, stream>>>(Ab, Wot, (float*)d_out,
                                              (int)BS, D_MODEL, D_MODEL);
}
